// Model_74354473828432
// MI455X (gfx1250) — compile-verified, matrix operand order checked by reading, not run on hardware
//
#include <hip/hip_runtime.h>
#include <cstdint>
#include <cstddef>

#define N_NODES 20000
#define N_EDGES 320000
#define DIM_IN  128
#define HEADS   6
#define CH      64
#define HCDIM   (HEADS * CH)   // 384
#define NCLS    16
#define NGRAPH  64
#define TOT_E   (N_EDGES + N_NODES)  // edges + self loops

#define MT 128   // block M tile
#define NT 64    // block N tile
#define KT 32    // K step (WMMA K)
#define LDSPAD 4 // row stride KT+4 = 36 halves -> 18-bank stride, conflict free

typedef __attribute__((ext_vector_type(16))) __bf16 v16bf;
typedef __attribute__((ext_vector_type(8)))  float  v8f;

__device__ __forceinline__ __bf16 f2bf(float f) {
  unsigned u = __builtin_bit_cast(unsigned, f);
  unsigned r = u + 0x7FFFu + ((u >> 16) & 1u);   // round-to-nearest-even
  unsigned short hs = (unsigned short)(r >> 16);
  return __builtin_bit_cast(__bf16, hs);
}

// ---------------------------------------------------------------------------
// Tiled WMMA GEMM: Cout[M x Nout] = A[M x K] @ W[K x Nout] (+ bias)
// 256 threads = 8 waves; block tile 128x64; wave w computes rows [16w,16w+16)
// across all 4 column subtiles, reusing its A fragment for 4 WMMAs/K-step.
// A and W staged through LDS as bf16 with coalesced b128 global loads.
// Requires: K % 32 == 0, Nout % 64 == 0. M arbitrary (loads clamped,
// stores guarded -- EXEC stays all-1s around the WMMAs).
// ---------------------------------------------------------------------------
__global__ __launch_bounds__(256) void gemm_bf16_wmma_tiled(
    const float* __restrict__ A, const float* __restrict__ W,
    const float* __restrict__ bias, float* __restrict__ Cout,
    int M, int K, int Nout) {
  __shared__ __bf16 As[MT][KT + LDSPAD];
  __shared__ __bf16 Ws[NT][KT + LDSPAD];   // transposed: Ws[col][k]

  const int tid   = threadIdx.x;
  const int wave  = tid >> 5;
  const int lane  = tid & 31;
  const int mlane = lane & 15;
  const int half  = lane >> 4;
  const int rowBase = blockIdx.x * MT;
  const int colBase = blockIdx.y * NT;

  v8f acc[4] = {};

  for (int kk = 0; kk < K; kk += KT) {
    // prefetch next K-step of W into cache (global_prefetch_b8 path)
    if (kk + KT < K)
      __builtin_prefetch(W + (size_t)(kk + KT + (tid >> 4)) * Nout + colBase + (tid & 15) * 4, 0, 3);

    // ---- stage A (MT x KT) -> LDS bf16, coalesced float4 loads ----
    {
      const int r0  = tid >> 3;   // 0..31
      const int vec = tid & 7;    // k offset = vec*4
#pragma unroll
      for (int p = 0; p < 4; ++p) {
        const int r = p * 32 + r0;
        int gr = rowBase + r; if (gr > M - 1) gr = M - 1;   // clamp (no divergence)
        const float4 f = *(const float4*)(A + (size_t)gr * K + kk + vec * 4);
        As[r][vec * 4 + 0] = f2bf(f.x);
        As[r][vec * 4 + 1] = f2bf(f.y);
        As[r][vec * 4 + 2] = f2bf(f.z);
        As[r][vec * 4 + 3] = f2bf(f.w);
      }
    }
    // ---- stage W (KT x NT) -> LDS bf16 transposed ----
    {
      const int kr0 = tid >> 4;   // 0..15
      const int vec = tid & 15;   // cols vec*4..vec*4+3
#pragma unroll
      for (int p = 0; p < 2; ++p) {
        const int kr = p * 16 + kr0;
        const float4 f = *(const float4*)(W + (size_t)(kk + kr) * Nout + colBase + vec * 4);
        Ws[vec * 4 + 0][kr] = f2bf(f.x);
        Ws[vec * 4 + 1][kr] = f2bf(f.y);
        Ws[vec * 4 + 2][kr] = f2bf(f.z);
        Ws[vec * 4 + 3][kr] = f2bf(f.w);
      }
    }
    __syncthreads();

    // ---- A fragment for this wave (16-bit A 16x32 layout) ----
    v16bf a;
    const int arow = wave * 16 + mlane;
#pragma unroll
    for (int v = 0; v < 8; ++v) {
      const int kb = half * 8 + ((v < 4) ? 2 * v : 16 + 2 * (v - 4));
      a[2 * v]     = As[arow][kb];
      a[2 * v + 1] = As[arow][kb + 1];
    }
    // ---- 4 column subtiles reuse the A fragment ----
#pragma unroll
    for (int nt = 0; nt < 4; ++nt) {
      v16bf b;
      const int bcol = nt * 16 + mlane;
#pragma unroll
      for (int v = 0; v < 8; ++v) {
        const int kb = half * 8 + ((v < 4) ? 2 * v : 16 + 2 * (v - 4));
        b[2 * v]     = Ws[bcol][kb];
        b[2 * v + 1] = Ws[bcol][kb + 1];
      }
      acc[nt] = __builtin_amdgcn_wmma_f32_16x16x32_bf16(
          false, a, false, b, (short)0, acc[nt], false, false);
    }
    __syncthreads();
  }

#pragma unroll
  for (int nt = 0; nt < 4; ++nt) {
    const int col = colBase + nt * 16 + mlane;
    const float bv = bias ? bias[col] : 0.0f;
#pragma unroll
    for (int v = 0; v < 8; ++v) {
      const int r = rowBase + wave * 16 + v + 8 * half;
      if (r < M) Cout[(size_t)r * Nout + col] = acc[nt][v] + bv;
    }
  }
}

// ---------------------------------------------------------------------------
// Simple one-wave-per-tile WMMA GEMM for small Nout (d2: 64x64 @ 64x16).
// grid = (M/16, Nout/16), block = 32. Dims must be multiples of 16/32.
// ---------------------------------------------------------------------------
__global__ void gemm_bf16_wmma(const float* __restrict__ A,
                               const float* __restrict__ W,
                               const float* __restrict__ bias,
                               float* __restrict__ Cout,
                               int M, int K, int Nout) {
  const int lane  = threadIdx.x & 31;
  const int mlane = lane & 15;
  const int half  = lane >> 4;
  const int row   = blockIdx.x * 16 + mlane;
  const int col   = blockIdx.y * 16 + mlane;

  v8f acc = {};
  for (int kk = 0; kk < K; kk += 32) {
    v16bf a, b;
#pragma unroll
    for (int v = 0; v < 8; ++v) {
      const int kbase = kk + half * 8 + ((v < 4) ? (2 * v) : (16 + 2 * (v - 4)));
      a[2 * v]     = f2bf(A[(size_t)row * K + kbase]);
      a[2 * v + 1] = f2bf(A[(size_t)row * K + kbase + 1]);
      b[2 * v]     = f2bf(W[(size_t)kbase * Nout + col]);
      b[2 * v + 1] = f2bf(W[(size_t)(kbase + 1) * Nout + col]);
    }
    acc = __builtin_amdgcn_wmma_f32_16x16x32_bf16(
        false, a, false, b, (short)0, acc, false, false);
  }
  const float bv = bias ? bias[col] : 0.0f;
#pragma unroll
  for (int v = 0; v < 8; ++v) {
    const int r = blockIdx.x * 16 + v + 8 * half;
    Cout[(size_t)r * Nout + col] = acc[v] + bv;
  }
}

// ---------------------------------------------------------------------------
// Edge attention logits: one wave per (edge, head).
// ---------------------------------------------------------------------------
__device__ __forceinline__ unsigned enc_f32(float f) {
  unsigned u = __builtin_bit_cast(unsigned, f);
  return (u & 0x80000000u) ? ~u : (u | 0x80000000u);
}
__device__ __forceinline__ float dec_f32(unsigned u) {
  unsigned b = (u & 0x80000000u) ? (u ^ 0x80000000u) : ~u;
  return __builtin_bit_cast(float, b);
}

__global__ void edge_logits(const float* __restrict__ xl,
                            const float* __restrict__ xr,
                            const float* __restrict__ att,
                            const int* __restrict__ ei,
                            float* __restrict__ ebuf,
                            unsigned* __restrict__ maxenc,
                            int heads, int outc) {
  const int warp = (int)((blockIdx.x * blockDim.x + threadIdx.x) >> 5);
  const int lane = threadIdx.x & 31;
  const int total = TOT_E * heads;
  if (warp >= total) return;
  const int eid = warp / heads;
  const int h   = warp - eid * heads;
  const int src = (eid < N_EDGES) ? ei[eid]            : (eid - N_EDGES);
  const int dst = (eid < N_EDGES) ? ei[N_EDGES + eid]  : (eid - N_EDGES);
  const int stride = heads * outc;
  const float* pl = xl + (size_t)src * stride + h * outc;
  const float* pr = xr + (size_t)dst * stride + h * outc;
  const float* pa = att + (size_t)h * outc;
  float s = 0.0f;
  for (int c = lane; c < outc; c += 32) {
    float v = pl[c] + pr[c];
    v = (v > 0.0f) ? v : 0.2f * v;
    s += v * pa[c];
  }
#pragma unroll
  for (int off = 16; off > 0; off >>= 1) s += __shfl_xor(s, off, 32);
  if (lane == 0) {
    ebuf[(size_t)eid * heads + h] = s;
    atomicMax(maxenc + (size_t)dst * heads + h, enc_f32(s));
  }
}

__global__ void edge_exp(const int* __restrict__ ei,
                         float* __restrict__ ebuf,
                         const unsigned* __restrict__ maxenc,
                         float* __restrict__ denom, int heads) {
  const int idx = blockIdx.x * blockDim.x + threadIdx.x;
  const int total = TOT_E * heads;
  if (idx >= total) return;
  const int eid = idx / heads;
  const int h   = idx - eid * heads;
  const int dst = (eid < N_EDGES) ? ei[N_EDGES + eid] : (eid - N_EDGES);
  const float m = dec_f32(maxenc[(size_t)dst * heads + h]);
  const float a = __expf(ebuf[idx] - m);
  ebuf[idx] = a;
  atomicAdd(denom + (size_t)dst * heads + h, a);
}

__global__ void edge_scatter(const int* __restrict__ ei,
                             const float* __restrict__ ebuf,
                             const float* __restrict__ denom,
                             const float* __restrict__ xl,
                             float* __restrict__ acc,
                             int heads, int outc) {
  __shared__ int s_src, s_dst;
  __shared__ float s_alpha[HEADS];
  const int eid = blockIdx.x;
  const int tid = threadIdx.x;
  if (tid == 0) {
    s_src = (eid < N_EDGES) ? ei[eid]           : (eid - N_EDGES);
    s_dst = (eid < N_EDGES) ? ei[N_EDGES + eid] : (eid - N_EDGES);
  }
  __syncthreads();
  if (tid < heads) {
    const float a = ebuf[(size_t)eid * heads + tid];
    const float d = denom[(size_t)s_dst * heads + tid];
    s_alpha[tid] = a / (d + 1e-16f);
  }
  __syncthreads();
  const int h = tid / outc;
  const int stride = heads * outc;
  const float v = xl[(size_t)s_src * stride + tid] * s_alpha[h];
  atomicAdd(acc + (size_t)s_dst * stride + tid, v);
}

__global__ void epilogue_elu(const float* __restrict__ acc,
                             const float* __restrict__ bias,
                             const float* __restrict__ xres,
                             float* __restrict__ out, int stride) {
  const size_t i = (size_t)blockIdx.x * blockDim.x + threadIdx.x;
  const size_t total = (size_t)N_NODES * stride;
  if (i >= total) return;
  const int c = (int)(i % stride);
  const float v = acc[i] + bias[c];
  const float e = (v > 0.0f) ? v : (__expf(v) - 1.0f);
  out[i] = e + xres[i];
}

__global__ void pool_sum(const float* __restrict__ h,
                         const int* __restrict__ batch,
                         float* __restrict__ pooled) {
  const int i = blockIdx.x * blockDim.x + threadIdx.x;
  if (i >= N_NODES * CH) return;
  const int n = i / CH, c = i - n * CH;
  atomicAdd(pooled + (size_t)batch[n] * CH + c, h[i]);
}

__global__ void bn_relu(float* __restrict__ z,
                        const float* __restrict__ g, const float* __restrict__ b,
                        const float* __restrict__ m, const float* __restrict__ v) {
  const int i = blockIdx.x * blockDim.x + threadIdx.x;
  if (i >= NGRAPH * CH) return;
  const int c = i % CH;
  float val = (z[i] - m[c]) * rsqrtf(v[c] + 1e-5f) * g[c] + b[c];
  z[i] = val > 0.0f ? val : 0.0f;
}

// ---------------------------------------------------------------------------
extern "C" void kernel_launch(void* const* d_in, const int* in_sizes, int n_in,
                              void* d_out, int out_size, void* d_ws, size_t ws_size,
                              hipStream_t stream) {
  const float* x       = (const float*)d_in[0];
  const int*   ei      = (const int*)d_in[1];
  const int*   batch   = (const int*)d_in[2];
  const float* inp_w   = (const float*)d_in[3];
  const float* inp_b   = (const float*)d_in[4];
  const float* l0_wl   = (const float*)d_in[5];
  const float* l0_bl   = (const float*)d_in[6];
  const float* l0_wr   = (const float*)d_in[7];
  const float* l0_br   = (const float*)d_in[8];
  const float* l0_att  = (const float*)d_in[9];
  const float* l0_bias = (const float*)d_in[10];
  const float* l1_wl   = (const float*)d_in[11];
  const float* l1_bl   = (const float*)d_in[12];
  const float* l1_wr   = (const float*)d_in[13];
  const float* l1_br   = (const float*)d_in[14];
  const float* l1_att  = (const float*)d_in[15];
  const float* l1_bias = (const float*)d_in[16];
  const float* l2_wl   = (const float*)d_in[17];
  const float* l2_bl   = (const float*)d_in[18];
  const float* l2_wr   = (const float*)d_in[19];
  const float* l2_br   = (const float*)d_in[20];
  const float* l2_att  = (const float*)d_in[21];
  const float* l2_bias = (const float*)d_in[22];
  const float* res_w   = (const float*)d_in[23];
  const float* d1_w    = (const float*)d_in[24];
  const float* d1_b    = (const float*)d_in[25];
  const float* bn_g    = (const float*)d_in[26];
  const float* bn_b    = (const float*)d_in[27];
  const float* bn_m    = (const float*)d_in[28];
  const float* bn_v    = (const float*)d_in[29];
  const float* d2_w    = (const float*)d_in[30];
  const float* d2_b    = (const float*)d_in[31];

  // Workspace layout (floats)
  float* ws = (float*)d_ws;
  const size_t NHC = (size_t)N_NODES * HCDIM;
  float*    XL     = ws;                          // N x HC
  float*    XR     = XL + NHC;                    // N x HC
  float*    ACC    = XR + NHC;                    // N x HC
  float*    HA     = ACC + NHC;                   // N x HC
  float*    HB     = HA + NHC;                    // N x HC
  float*    EBUF   = HB + NHC;                    // TOT_E x HEADS
  float*    DEN    = EBUF + (size_t)TOT_E * HEADS;// N x HEADS
  unsigned* MAXE   = (unsigned*)(DEN + (size_t)N_NODES * HEADS); // N x HEADS
  float*    POOLED = (float*)(MAXE + (size_t)N_NODES * HEADS);   // NG x CH
  float*    Z1     = POOLED + (size_t)NGRAPH * CH;               // NG x CH

  float* out_h = (float*)d_out;                    // N x CH
  float* out_z = out_h + (size_t)N_NODES * CH;     // NG x NCLS

  const int gmx = (N_NODES + MT - 1) / MT;         // 157
  const dim3 blk(256);
  const int logit_blocks6 = (TOT_E * HEADS + 7) / 8;
  const int logit_blocks1 = (TOT_E * 1 + 7) / 8;
  const int exp_blocks6   = (TOT_E * HEADS + 255) / 256;
  const int exp_blocks1   = (TOT_E * 1 + 255) / 256;
  const int epi_blocks384 = (int)(((size_t)N_NODES * HCDIM + 255) / 256);
  const int epi_blocks64  = (int)(((size_t)N_NODES * CH + 255) / 256);

  // ===== Layer 0: input 128 -> (6,64), concat =====
  gemm_bf16_wmma_tiled<<<dim3(gmx, HCDIM / NT), blk, 0, stream>>>(
      x, inp_w, inp_b, HA, N_NODES, DIM_IN, HCDIM);           // x_res
  gemm_bf16_wmma_tiled<<<dim3(gmx, HCDIM / NT), blk, 0, stream>>>(
      x, l0_wl, l0_bl, XL, N_NODES, DIM_IN, HCDIM);
  gemm_bf16_wmma_tiled<<<dim3(gmx, HCDIM / NT), blk, 0, stream>>>(
      x, l0_wr, l0_br, XR, N_NODES, DIM_IN, HCDIM);
  hipMemsetAsync(ACC, 0, NHC * sizeof(float), stream);
  hipMemsetAsync(DEN, 0, (size_t)N_NODES * HEADS * sizeof(float), stream);
  hipMemsetAsync(MAXE, 0, (size_t)N_NODES * HEADS * sizeof(unsigned), stream);
  edge_logits<<<logit_blocks6, 256, 0, stream>>>(XL, XR, l0_att, ei, EBUF, MAXE, HEADS, CH);
  edge_exp<<<exp_blocks6, 256, 0, stream>>>(ei, EBUF, MAXE, DEN, HEADS);
  edge_scatter<<<TOT_E, HCDIM, 0, stream>>>(ei, EBUF, DEN, XL, ACC, HEADS, CH);
  epilogue_elu<<<epi_blocks384, 256, 0, stream>>>(ACC, l0_bias, HA, HA, HCDIM);

  // ===== Layer 1: 384 -> (6,64), concat =====
  gemm_bf16_wmma_tiled<<<dim3(gmx, HCDIM / NT), blk, 0, stream>>>(
      HA, l1_wl, l1_bl, XL, N_NODES, HCDIM, HCDIM);
  gemm_bf16_wmma_tiled<<<dim3(gmx, HCDIM / NT), blk, 0, stream>>>(
      HA, l1_wr, l1_br, XR, N_NODES, HCDIM, HCDIM);
  hipMemsetAsync(ACC, 0, NHC * sizeof(float), stream);
  hipMemsetAsync(DEN, 0, (size_t)N_NODES * HEADS * sizeof(float), stream);
  hipMemsetAsync(MAXE, 0, (size_t)N_NODES * HEADS * sizeof(unsigned), stream);
  edge_logits<<<logit_blocks6, 256, 0, stream>>>(XL, XR, l1_att, ei, EBUF, MAXE, HEADS, CH);
  edge_exp<<<exp_blocks6, 256, 0, stream>>>(ei, EBUF, MAXE, DEN, HEADS);
  edge_scatter<<<TOT_E, HCDIM, 0, stream>>>(ei, EBUF, DEN, XL, ACC, HEADS, CH);
  epilogue_elu<<<epi_blocks384, 256, 0, stream>>>(ACC, l1_bias, HA, HB, HCDIM);

  // ===== Layer 2: 384 -> (1,64); residual proj 384->64 =====
  gemm_bf16_wmma_tiled<<<dim3(gmx, CH / NT), blk, 0, stream>>>(
      HB, l2_wl, l2_bl, XL, N_NODES, HCDIM, CH);
  gemm_bf16_wmma_tiled<<<dim3(gmx, CH / NT), blk, 0, stream>>>(
      HB, l2_wr, l2_br, XR, N_NODES, HCDIM, CH);
  gemm_bf16_wmma_tiled<<<dim3(gmx, CH / NT), blk, 0, stream>>>(
      HB, res_w, nullptr, out_h, N_NODES, HCDIM, CH);
  hipMemsetAsync(ACC, 0, (size_t)N_NODES * CH * sizeof(float), stream);
  hipMemsetAsync(DEN, 0, (size_t)N_NODES * sizeof(float), stream);
  hipMemsetAsync(MAXE, 0, (size_t)N_NODES * sizeof(unsigned), stream);
  edge_logits<<<logit_blocks1, 256, 0, stream>>>(XL, XR, l2_att, ei, EBUF, MAXE, 1, CH);
  edge_exp<<<exp_blocks1, 256, 0, stream>>>(ei, EBUF, MAXE, DEN, 1);
  edge_scatter<<<TOT_E, CH, 0, stream>>>(ei, EBUF, DEN, XL, ACC, 1, CH);
  epilogue_elu<<<epi_blocks64, 256, 0, stream>>>(ACC, l2_bias, out_h, out_h, CH);

  // ===== Pool + dense head =====
  hipMemsetAsync(POOLED, 0, (size_t)NGRAPH * CH * sizeof(float), stream);
  pool_sum<<<(N_NODES * CH + 255) / 256, 256, 0, stream>>>(out_h, batch, POOLED);
  gemm_bf16_wmma_tiled<<<dim3(1, CH / NT), blk, 0, stream>>>(
      POOLED, d1_w, d1_b, Z1, NGRAPH, CH, CH);
  bn_relu<<<(NGRAPH * CH + 255) / 256, 256, 0, stream>>>(Z1, bn_g, bn_b, bn_m, bn_v);
  gemm_bf16_wmma<<<dim3(NGRAPH / 16, NCLS / 16), dim3(32), 0, stream>>>(
      Z1, d2_w, d2_b, out_z, NGRAPH, CH, NCLS);
}
